// CareerGraphGNN_38912403702363
// MI455X (gfx1250) — matrix operand order repaired and checked by reading, not compiled, not run on hardware
//
#include <hip/hip_runtime.h>

// ---------------------------------------------------------------------------
// CDNA5 (gfx1250) GCN forward: bf16 WMMA GEMMs + edge scatter aggregation
//  - W pre-packed to WMMA B-fragment-major layout (coalesced b128 loads)
//  - 4 N-tiles per wave, 2-stage software-pipelined k-loop (ping-pong regs)
//  - explicit global-address-space loads (global_load_b128, no flat/DScnt)
// ---------------------------------------------------------------------------

typedef __attribute__((ext_vector_type(16))) __bf16 v16bf;
typedef __attribute__((ext_vector_type(8)))  float  v8f;
typedef __attribute__((ext_vector_type(4)))  unsigned int v4u;
typedef __attribute__((ext_vector_type(2)))  unsigned int v2u;

#define GAS __attribute__((address_space(1)))

__device__ __forceinline__ v4u ldg_q(const unsigned short* p) {
    return *(const GAS v4u*)(unsigned long long)(uintptr_t)p;
}
__device__ __forceinline__ v2u ldg_d(const unsigned short* p) {
    return *(const GAS v2u*)(unsigned long long)(uintptr_t)p;
}
__device__ __forceinline__ void stg_h(unsigned short* p, unsigned short v) {
    *(GAS unsigned short*)(unsigned long long)(uintptr_t)p = v;
}

__device__ __forceinline__ unsigned short f2bf(float x) {
    unsigned int u = __float_as_uint(x);
    unsigned int r = u + 0x7FFFu + ((u >> 16) & 1u);   // round-to-nearest-even
    return (unsigned short)(r >> 16);
}
__device__ __forceinline__ float bf2f(unsigned short s) {
    return __uint_as_float(((unsigned int)s) << 16);
}

// ---- elementwise helpers ---------------------------------------------------

__global__ void k_f32_to_bf16(const float* __restrict__ in,
                              unsigned short* __restrict__ out, size_t n) {
    size_t i = (size_t)blockIdx.x * blockDim.x + threadIdx.x;
    if (i < n) out[i] = f2bf(in[i]);
}

__global__ void k_fill(float* __restrict__ p, float v, size_t n) {
    size_t i = (size_t)blockIdx.x * blockDim.x + threadIdx.x;
    if (i < n) p[i] = v;
}

__global__ void k_deg(const int* __restrict__ dst, float* __restrict__ deg, int E) {
    int i = blockIdx.x * blockDim.x + threadIdx.x;
    if (i < E) atomicAdd(&deg[dst[i]], 1.0f);
}

__global__ void k_rsqrt(const float* __restrict__ deg, float* __restrict__ dinv, int n) {
    int i = blockIdx.x * blockDim.x + threadIdx.x;
    if (i < n) {
        float d = deg[i];
        dinv[i] = (d > 0.f) ? rsqrtf(d) : 0.f;
    }
}

// ---- pack W (f32, KxN row-major) into WMMA B-fragment-major bf16 ----------
// Fragment (tn, kb32): 32 lanes x 16 bf16, lane-contiguous (32B per lane).
// Element e of lane (g = lane>>4, n = lane&15) is W[kb32*32 + (e<8 ? 8g+e : 16+8g+e-8)][tn*16+n].

__global__ void k_pack_B(const float* __restrict__ W,
                         unsigned short* __restrict__ Bp, int K, int N) {
    size_t i = (size_t)blockIdx.x * blockDim.x + threadIdx.x;
    size_t total = (size_t)K * N;
    if (i >= total) return;
    int e    = (int)(i & 15);
    int lane = (int)((i >> 4) & 31);
    size_t rest = i >> 9;
    int KB   = K >> 5;
    int kb32 = (int)(rest % KB);
    int tn   = (int)(rest / KB);
    int g = lane >> 4, n = lane & 15;
    int k   = kb32 * 32 + (e < 8 ? 8 * g + e : 16 + 8 * g + (e - 8));
    int col = tn * 16 + n;
    Bp[i] = f2bf(W[(size_t)k * N + col]);
}

// ---- WMMA bf16 GEMM: C[M,N] = A[M,K] @ W[K,N] ------------------------------
// One wave32 per 16x64 C strip (4 N-tiles), 4 waves per block.
// 2-stage pipelined k-loop: loads for stage k+1 in flight during WMMAs of k.

#define WMMA_BF16(a, b, c) \
    __builtin_amdgcn_wmma_f32_16x16x32_bf16(false, (a), false, (b), (short)0, (c), false, false)

__global__ void __launch_bounds__(128)
k_gemm_bf16(const unsigned short* __restrict__ A,
            const unsigned short* __restrict__ Bp,
            unsigned short* __restrict__ C,
            int K, int N, int groupsN, int totalWaves) {
    const int wave = threadIdx.x >> 5;
    const int lane = threadIdx.x & 31;
    const int wid  = blockIdx.x * 4 + wave;
    if (wid >= totalWaves) return;          // wave-uniform: no partial-EXEC WMMA

    const int tm = wid / groupsN;
    const int tg = wid % groupsN;           // group of 4 consecutive N-tiles
    const int g  = lane >> 4;
    const int mn = lane & 15;
    const int KB = K >> 5;                  // even (24 / 8 / 8), >= 2

    // A fragment pointer: chunks at +0 and +16 ushorts; advance 32/step.
    const unsigned short* ap = A + (size_t)(tm * 16 + mn) * (size_t)K + 8 * g;
    // B fragment pointers (one per N-tile): 16 contiguous ushorts per lane,
    // stride 512 ushorts per k-step.
    const unsigned short* pb[4];
#pragma unroll
    for (int j = 0; j < 4; ++j)
        pb[j] = Bp + ((size_t)((tg * 4 + j) * KB) * 32 + lane) * 16;

    v8f acc[4] = {{}, {}, {}, {}};
    union Frag { v4u q[2]; v16bf v; };
    Frag af[2], bf[2][4];

    // prologue: stage 0 -> buffer 0
    af[0].q[0] = ldg_q(ap);
    af[0].q[1] = ldg_q(ap + 16);
#pragma unroll
    for (int j = 0; j < 4; ++j) {
        bf[0][j].q[0] = ldg_q(pb[j]);
        bf[0][j].q[1] = ldg_q(pb[j] + 8);
    }

    for (int kb32 = 0; kb32 < KB - 2; kb32 += 2) {
        // stage kb32+1 -> buffer 1 (in flight during buffer-0 WMMAs)
        af[1].q[0] = ldg_q(ap + 32);
        af[1].q[1] = ldg_q(ap + 48);
#pragma unroll
        for (int j = 0; j < 4; ++j) {
            bf[1][j].q[0] = ldg_q(pb[j] + 512);
            bf[1][j].q[1] = ldg_q(pb[j] + 520);
        }
#pragma unroll
        for (int j = 0; j < 4; ++j) acc[j] = WMMA_BF16(af[0].v, bf[0][j].v, acc[j]);

        ap += 64;
#pragma unroll
        for (int j = 0; j < 4; ++j) pb[j] += 1024;

        // stage kb32+2 -> buffer 0 (in flight during buffer-1 WMMAs)
        af[0].q[0] = ldg_q(ap);
        af[0].q[1] = ldg_q(ap + 16);
#pragma unroll
        for (int j = 0; j < 4; ++j) {
            bf[0][j].q[0] = ldg_q(pb[j]);
            bf[0][j].q[1] = ldg_q(pb[j] + 8);
        }
#pragma unroll
        for (int j = 0; j < 4; ++j) acc[j] = WMMA_BF16(af[1].v, bf[1][j].v, acc[j]);
    }

    // epilogue: stages KB-2 (already in buffer 0) and KB-1
    af[1].q[0] = ldg_q(ap + 32);
    af[1].q[1] = ldg_q(ap + 48);
#pragma unroll
    for (int j = 0; j < 4; ++j) {
        bf[1][j].q[0] = ldg_q(pb[j] + 512);
        bf[1][j].q[1] = ldg_q(pb[j] + 520);
    }
#pragma unroll
    for (int j = 0; j < 4; ++j) acc[j] = WMMA_BF16(af[0].v, bf[0][j].v, acc[j]);
#pragma unroll
    for (int j = 0; j < 4; ++j) acc[j] = WMMA_BF16(af[1].v, bf[1][j].v, acc[j]);

    // C layout: VGPR r -> row (tm*16 + r + 8g), col = tn*16 + n
#pragma unroll
    for (int j = 0; j < 4; ++j) {
        const int tn = tg * 4 + j;
        unsigned short* cp = C + (size_t)(tm * 16 + 8 * g) * (size_t)N + tn * 16 + mn;
#pragma unroll
        for (int r = 0; r < 8; ++r) stg_h(cp + (size_t)r * N, f2bf(acc[j][r]));
    }
}

// ---- normalized edge aggregation: out[dst] += dinv[s]*dinv[d]*H[src] -------
// One block per edge (blocks >= E are self-loops), one thread per 4 features
// (one b64 bf16x4 gather + 4 f32 atomics; destination rows live in L2).

__global__ void k_aggregate(const unsigned short* __restrict__ H,
                            const int* __restrict__ src,
                            const int* __restrict__ dst,
                            const float* __restrict__ dinv,
                            float* __restrict__ out,
                            int E, int F) {
    int e = blockIdx.x;
    int t = threadIdx.x;            // blockDim = F/4
    int s, d;
    if (e < E) { s = src[e]; d = dst[e]; }
    else       { s = e - E; d = s; }                 // self-loop
    float nrm = dinv[s] * dinv[d];
    int f = t * 4;
    v2u h4 = ldg_d(H + (size_t)s * F + f);
    float* op = out + (size_t)d * F + f;
    atomicAdd(op + 0, bf2f((unsigned short)(h4[0] & 0xFFFF)) * nrm);
    atomicAdd(op + 1, bf2f((unsigned short)(h4[0] >> 16))    * nrm);
    atomicAdd(op + 2, bf2f((unsigned short)(h4[1] & 0xFFFF)) * nrm);
    atomicAdd(op + 3, bf2f((unsigned short)(h4[1] >> 16))    * nrm);
}

__global__ void k_bias_relu_bf16(const float* __restrict__ agg,
                                 const float* __restrict__ b,
                                 unsigned short* __restrict__ out,
                                 int F, size_t total) {
    size_t i = (size_t)blockIdx.x * blockDim.x + threadIdx.x;
    if (i < total) {
        float v = agg[i] + b[i % (size_t)F];
        out[i] = f2bf(v > 0.f ? v : 0.f);
    }
}

__global__ void k_bias_add(float* __restrict__ agg, const float* __restrict__ b,
                           int F, size_t total) {
    size_t i = (size_t)blockIdx.x * blockDim.x + threadIdx.x;
    if (i < total) agg[i] += b[i % (size_t)F];
}

// ---- mean over nodes: one block per feature --------------------------------

__global__ void k_mean(const float* __restrict__ h,
                       float* __restrict__ outA, float* __restrict__ outB,
                       int nNodes, int F) {
    __shared__ float s[256];
    int f = blockIdx.x;
    float acc = 0.f;
    for (int nrow = threadIdx.x; nrow < nNodes; nrow += blockDim.x)
        acc += h[(size_t)nrow * F + f];
    s[threadIdx.x] = acc;
    __syncthreads();
    for (int off = 128; off > 0; off >>= 1) {
        if ((int)threadIdx.x < off) s[threadIdx.x] += s[threadIdx.x + off];
        __syncthreads();
    }
    if (threadIdx.x == 0) {
        float m = s[0] / (float)nNodes;
        outA[f] = m;
        outB[f] = m;
    }
}

// ---- three sigmoid heads on the pooled embedding ---------------------------

__global__ void k_heads(const float* __restrict__ emb,
                        const float* __restrict__ Wl, const float* __restrict__ bl,
                        const float* __restrict__ Wi, const float* __restrict__ bi,
                        const float* __restrict__ Wg, const float* __restrict__ bg,
                        float* __restrict__ out, int F) {
    __shared__ float sl[128], si[128], sg[128];
    int t = threadIdx.x;
    float e = emb[t];
    sl[t] = e * Wl[t];
    si[t] = e * Wi[t];
    sg[t] = e * Wg[t];
    __syncthreads();
    for (int off = 64; off > 0; off >>= 1) {
        if (t < off) { sl[t] += sl[t + off]; si[t] += si[t + off]; sg[t] += sg[t + off]; }
        __syncthreads();
    }
    if (t == 0) {
        out[F + 0] = 1.f / (1.f + __expf(-(sl[0] + bl[0])));
        out[F + 1] = 1.f / (1.f + __expf(-(si[0] + bi[0])));
        out[F + 2] = 1.f / (1.f + __expf(-(sg[0] + bg[0])));
    }
}

// ---------------------------------------------------------------------------

extern "C" void kernel_launch(void* const* d_in, const int* in_sizes, int n_in,
                              void* d_out, int out_size, void* d_ws, size_t ws_size,
                              hipStream_t stream) {
    const float* x  = (const float*)d_in[0];
    const int* eidx = (const int*)d_in[1];
    const float* W1 = (const float*)d_in[2];  const float* b1 = (const float*)d_in[3];
    const float* W2 = (const float*)d_in[4];  const float* b2 = (const float*)d_in[5];
    const float* W3 = (const float*)d_in[6];  const float* b3 = (const float*)d_in[7];
    const float* Wl = (const float*)d_in[8];  const float* bl = (const float*)d_in[9];
    const float* Wi = (const float*)d_in[10]; const float* bi = (const float*)d_in[11];
    const float* Wg = (const float*)d_in[12]; const float* bg = (const float*)d_in[13];

    const int Fin = 768, Hd = 256, Od = 128;
    const int Nn = in_sizes[0] / Fin;       // 50000 (multiple of 16)
    const int E  = in_sizes[1] / 2;         // 800000
    const int* srcE = eidx;
    const int* dstE = eidx + E;

    auto a256 = [](size_t v) { return (v + 255) & ~(size_t)255; };
    char* ws = (char*)d_ws;
    size_t off = 0;
    unsigned short* Abf = (unsigned short*)(ws + off); off = a256(off + (size_t)Nn * Fin * 2); // activations (bf16)
    unsigned short* Hbf = (unsigned short*)(ws + off); off = a256(off + (size_t)Nn * Hd * 2);  // GEMM out (bf16)
    float* AGG          = (float*)(ws + off);          off = a256(off + (size_t)Nn * Hd * 4);  // scatter accum (f32)
    unsigned short* Bp  = (unsigned short*)(ws + off); off = a256(off + (size_t)Fin * Hd * 2); // packed W (bf16)
    float* deg          = (float*)(ws + off);          off = a256(off + (size_t)Nn * 4);
    float* dinv         = (float*)(ws + off);          off = a256(off + (size_t)Nn * 4);
    float* emb          = (float*)(ws + off);          off = a256(off + 512);

    // ---- degrees (self-loop contributes 1) ----
    k_fill<<<(Nn + 255) / 256, 256, 0, stream>>>(deg, 1.0f, (size_t)Nn);
    k_deg<<<(E + 255) / 256, 256, 0, stream>>>(dstE, deg, E);
    k_rsqrt<<<(Nn + 255) / 256, 256, 0, stream>>>(deg, dinv, Nn);

    // ---- input -> bf16 ----
    {
        size_t n = (size_t)Nn * Fin;
        k_f32_to_bf16<<<(unsigned)((n + 255) / 256), 256, 0, stream>>>(x, Abf, n);
    }

    // ================= layer 1: 768 -> 256, ReLU =================
    {
        size_t n = (size_t)Fin * Hd;
        k_pack_B<<<(unsigned)((n + 255) / 256), 256, 0, stream>>>(W1, Bp, Fin, Hd);
        int groupsN    = Hd / 64;                          // 4
        int totalWaves = (Nn / 16) * groupsN;              // 12500
        k_gemm_bf16<<<(totalWaves + 3) / 4, 128, 0, stream>>>(Abf, Bp, Hbf, Fin, Hd,
                                                              groupsN, totalWaves);
        size_t na = (size_t)Nn * Hd;
        k_fill<<<(unsigned)((na + 255) / 256), 256, 0, stream>>>(AGG, 0.f, na);
        k_aggregate<<<E + Nn, Hd / 4, 0, stream>>>(Hbf, srcE, dstE, dinv, AGG, E, Hd);
        k_bias_relu_bf16<<<(unsigned)((na + 255) / 256), 256, 0, stream>>>(AGG, b1, Abf, Hd, na);
    }

    // ================= layer 2: 256 -> 256, ReLU =================
    {
        size_t n = (size_t)Hd * Hd;
        k_pack_B<<<(unsigned)((n + 255) / 256), 256, 0, stream>>>(W2, Bp, Hd, Hd);
        int groupsN    = Hd / 64;
        int totalWaves = (Nn / 16) * groupsN;
        k_gemm_bf16<<<(totalWaves + 3) / 4, 128, 0, stream>>>(Abf, Bp, Hbf, Hd, Hd,
                                                              groupsN, totalWaves);
        size_t na = (size_t)Nn * Hd;
        k_fill<<<(unsigned)((na + 255) / 256), 256, 0, stream>>>(AGG, 0.f, na);
        k_aggregate<<<E + Nn, Hd / 4, 0, stream>>>(Hbf, srcE, dstE, dinv, AGG, E, Hd);
        k_bias_relu_bf16<<<(unsigned)((na + 255) / 256), 256, 0, stream>>>(AGG, b2, Abf, Hd, na);
    }

    // ================= layer 3: 256 -> 128, no ReLU =================
    {
        size_t n = (size_t)Hd * Od;
        k_pack_B<<<(unsigned)((n + 255) / 256), 256, 0, stream>>>(W3, Bp, Hd, Od);
        int groupsN    = Od / 64;                          // 2
        int totalWaves = (Nn / 16) * groupsN;              // 6250
        k_gemm_bf16<<<(totalWaves + 3) / 4, 128, 0, stream>>>(Abf, Bp, Hbf, Hd, Od,
                                                              groupsN, totalWaves);
        size_t na = (size_t)Nn * Od;
        k_fill<<<(unsigned)((na + 255) / 256), 256, 0, stream>>>(AGG, 0.f, na);
        k_aggregate<<<E + Nn, Od / 4, 0, stream>>>(Hbf, srcE, dstE, dinv, AGG, E, Od);
        k_bias_add<<<(unsigned)((na + 255) / 256), 256, 0, stream>>>(AGG, b3, Od, na);
    }

    // ---- mean-pool + heads ----
    float* out = (float*)d_out;                 // [0..127]=emb, 128..130 = heads
    k_mean<<<Od, 256, 0, stream>>>(AGG, out, emb, Nn, Od);
    k_heads<<<1, Od, 0, stream>>>(emb, Wl, bl, Wi, bi, Wg, bg, out, Od);
}
